// Attention_7756710936846
// MI455X (gfx1250) — compile-verified, matrix operand order checked by reading, not run on hardware
//
#include <hip/hip_runtime.h>

// MI455X (gfx1250) multi-head attention: fp32 in/out, f16 WMMA internally.
//   k_qkv : qkv = x @ w_qkv           (f32 -> f16 tiles -> v_wmma f16, f32 acc)
//   k_attn: flash attention per (b,h) (QK^T and PV via v_wmma, online softmax,
//           double-buffered K/V staged with global_load_async_to_lds_b128)
//   k_out : out = ao @ w_out + b_out  (f16 A, f32 out)

typedef __attribute__((ext_vector_type(16))) _Float16 v16h;
typedef __attribute__((ext_vector_type(8)))  _Float16 v8h;
typedef __attribute__((ext_vector_type(8)))  float    v8f;
typedef __attribute__((ext_vector_type(4)))  float    v4f;

#define DIM    1024
#define SEQ    2048
#define BATCH  4
#define HEADS  16
#define DHEAD  64
#define NQKV   3072
#define M_TOT  (BATCH * SEQ)   // 8192

// Build one 16x32 f16 WMMA operand (A-layout, or B via [N][K]-transposed LDS).
// ISA 16-bit A layout: lanes 0-15 hold K {kb..kb+7, 16+kb..23+kb} with kb=0,
// lanes 16-31 the same with kb=8.
__device__ __forceinline__ v16h frag_row(const _Float16* rowp, int lane) {
  const int kb = (lane >> 4) << 3;   // 0 or 8
  v8h lo = *(const v8h*)(rowp + kb);
  v8h hi = *(const v8h*)(rowp + 16 + kb);
  v16h r;
#pragma unroll
  for (int i = 0; i < 8; ++i) { r[i] = lo[i]; r[8 + i] = hi[i]; }
  return r;
}

// LDS byte offset of a __shared__ object: flat LDS addresses truncate to the
// wave-relative LDS address (ISA 10.2 aperture mapping).
__device__ __forceinline__ unsigned lds_off(const void* p) {
  return (unsigned)(uintptr_t)p;
}

// 16B memory -> LDS async copy (per lane), tracked by ASYNCcnt.
__device__ __forceinline__ void async_copy16(unsigned lds_addr, const void* g) {
  unsigned long long ga = (unsigned long long)(uintptr_t)g;
  asm volatile("global_load_async_to_lds_b128 %0, %1, off"
               :: "v"(lds_addr), "v"(ga) : "memory");
}

// ---------------------------------------------------------------- GEMM1: qkv
// 256 thr = 8 waves (2x4), BM=64, BN=128, BK=32; each wave: 32x32 out tile.
__global__ __launch_bounds__(256) void k_qkv(const float* __restrict__ x,
                                             const float* __restrict__ wqkv,
                                             _Float16* __restrict__ qws,
                                             _Float16* __restrict__ kws,
                                             _Float16* __restrict__ vws) {
  __shared__ _Float16 As[64 * 40];    // [m][k], stride 40 (16B aligned rows)
  __shared__ _Float16 Bs[128 * 40];   // transposed: [n][k]
  const int t = threadIdx.x, lane = t & 31, wave = t >> 5;
  const int wm = wave >> 2, wn = wave & 3;
  const int m0 = blockIdx.y * 64, n0 = blockIdx.x * 128;
  v8f acc[2][2] = {};

  for (int k0 = 0; k0 < DIM; k0 += 32) {
    // prefetch next k-tile (global_prefetch_b8)
    if (k0 + 32 < DIM) {
      __builtin_prefetch(x + (size_t)(m0 + (t >> 3)) * DIM + k0 + 32 + (t & 7) * 4, 0, 1);
      __builtin_prefetch(wqkv + (size_t)(k0 + 32 + (t >> 5)) * NQKV + n0 + (t & 31) * 4, 0, 1);
    }
    // A tile: 64x32 fp32 -> f16 LDS (2 float4 per thread)
#pragma unroll
    for (int i = 0; i < 2; ++i) {
      int v = t + 256 * i;
      int row = v >> 3, c4 = (v & 7) * 4;
      v4f f = *(const v4f*)(x + (size_t)(m0 + row) * DIM + k0 + c4);
      _Float16* d = &As[row * 40 + c4];
      d[0] = (_Float16)f.x; d[1] = (_Float16)f.y;
      d[2] = (_Float16)f.z; d[3] = (_Float16)f.w;
    }
    // B tile: 32x128 fp32 -> transposed f16 LDS [n][k] (4 float4 per thread)
#pragma unroll
    for (int i = 0; i < 4; ++i) {
      int v = t + 256 * i;
      int kr = v >> 5, c4 = (v & 31) * 4;
      v4f f = *(const v4f*)(wqkv + (size_t)(k0 + kr) * NQKV + n0 + c4);
      Bs[(c4 + 0) * 40 + kr] = (_Float16)f.x;
      Bs[(c4 + 1) * 40 + kr] = (_Float16)f.y;
      Bs[(c4 + 2) * 40 + kr] = (_Float16)f.z;
      Bs[(c4 + 3) * 40 + kr] = (_Float16)f.w;
    }
    __syncthreads();
    v16h aF[2], bF[2];
#pragma unroll
    for (int mi = 0; mi < 2; ++mi)
      aF[mi] = frag_row(&As[(wm * 32 + mi * 16 + (lane & 15)) * 40], lane);
#pragma unroll
    for (int ni = 0; ni < 2; ++ni)
      bF[ni] = frag_row(&Bs[(wn * 32 + ni * 16 + (lane & 15)) * 40], lane);
#pragma unroll
    for (int mi = 0; mi < 2; ++mi)
#pragma unroll
      for (int ni = 0; ni < 2; ++ni)
        acc[mi][ni] = __builtin_amdgcn_wmma_f32_16x16x32_f16(
            false, aF[mi], false, bF[ni], (short)0, acc[mi][ni], false, false);
    __syncthreads();
  }

  // Epilogue: scatter q (x 1/sqrt(64)), k as [b,h,n,d]; v as [b,h,d,n].
#pragma unroll
  for (int mi = 0; mi < 2; ++mi)
#pragma unroll
    for (int ni = 0; ni < 2; ++ni) {
      int c = n0 + wn * 32 + ni * 16 + (lane & 15);
      int part = c >> 10;            // 0=q 1=k 2=v
      int inner = c & 1023;
      int h = inner >> 6, d = inner & 63;
#pragma unroll
      for (int g = 0; g < 8; ++g) {
        int r = m0 + wm * 32 + mi * 16 + ((lane < 16) ? g : g + 8);
        int b = r >> 11, n = r & 2047;
        float val = acc[mi][ni][g];
        if (part == 0)
          qws[(((size_t)(b * HEADS + h)) * SEQ + n) * DHEAD + d] =
              (_Float16)(val * 0.125f);
        else if (part == 1)
          kws[(((size_t)(b * HEADS + h)) * SEQ + n) * DHEAD + d] = (_Float16)val;
        else
          vws[(((size_t)(b * HEADS + h)) * DHEAD + d) * SEQ + n] = (_Float16)val;
      }
    }
}

// --------------------------------------------------------- flash attention
// 128 thr = 4 waves; block = 64 query rows, each wave owns 16 rows.
// K/V tiles double-buffered in LDS via async global->LDS copies (ASYNCcnt).
__global__ __launch_bounds__(128) void k_attn(const _Float16* __restrict__ qws,
                                              const _Float16* __restrict__ kws,
                                              const _Float16* __restrict__ vws,
                                              _Float16* __restrict__ ows) {
  __shared__ _Float16 Qs[64 * 72];             // [m][d]
  __shared__ _Float16 Ks[2][64 * 72];          // [j][d]  (B^T layout for QK^T)
  __shared__ _Float16 Vs[2][64 * 72];          // [d][j]  (B^T layout for PV)
  __shared__ _Float16 Ps[4][16 * 72];          // per-wave P tile [m][j]
  const int t = threadIdx.x, lane = t & 31, wave = t >> 5;
  const int bh = blockIdx.y;
  const int n0 = blockIdx.x * 64;
  const _Float16* qbase = qws + (size_t)bh * SEQ * DHEAD;
  const _Float16* kbase = kws + (size_t)bh * SEQ * DHEAD;
  const _Float16* vbase = vws + (size_t)bh * DHEAD * SEQ;

  // stage Q tile once (synchronous), keep A fragments in registers
#pragma unroll
  for (int i = 0; i < 4; ++i) {
    int v = t + 128 * i;
    int row = v >> 3, c8 = (v & 7) * 8;
    *(v8h*)&Qs[row * 72 + c8] =
        *(const v8h*)(qbase + (size_t)(n0 + row) * DHEAD + c8);
  }

  // prologue: async-stage K/V tile 0 into buffer 0 (8 async ops per wave)
#pragma unroll
  for (int i = 0; i < 4; ++i) {
    int v = t + 128 * i;
    int row = v >> 3, c8 = (v & 7) * 8;
    async_copy16(lds_off(&Ks[0][row * 72 + c8]),
                 kbase + (size_t)row * DHEAD + c8);
    async_copy16(lds_off(&Vs[0][row * 72 + c8]),
                 vbase + (size_t)row * SEQ + c8);
  }
  __syncthreads();   // Qs visible to all waves
  const int mrow = wave * 16 + (lane & 15);
  v16h qF[2];
  qF[0] = frag_row(&Qs[mrow * 72 + 0], lane);
  qF[1] = frag_row(&Qs[mrow * 72 + 32], lane);

  v8f o[4] = {};
  float mr[8], lr[8];
#pragma unroll
  for (int g = 0; g < 8; ++g) { mr[g] = -1e30f; lr[g] = 0.f; }
  const float L2E = 1.44269504088896340736f;

  for (int j0 = 0; j0 < SEQ; j0 += 64) {
    const int cur = (j0 >> 6) & 1, nxt = cur ^ 1;
    if (j0 + 64 < SEQ) {
      // issue next tile's async copies (buffer nxt was released by the
      // trailing barrier of the previous iteration)
      const int j1 = j0 + 64;
#pragma unroll
      for (int i = 0; i < 4; ++i) {
        int v = t + 128 * i;
        int row = v >> 3, c8 = (v & 7) * 8;
        async_copy16(lds_off(&Ks[nxt][row * 72 + c8]),
                     kbase + (size_t)(j1 + row) * DHEAD + c8);
        async_copy16(lds_off(&Vs[nxt][row * 72 + c8]),
                     vbase + (size_t)row * SEQ + j1 + c8);
      }
      // 8 newly issued may stay in flight; the previous tile's 8 (older,
      // async loads complete in order) must be done.
      asm volatile("s_wait_asynccnt 8" ::: "memory");
    } else {
      asm volatile("s_wait_asynccnt 0" ::: "memory");
    }
    __syncthreads();   // all waves' staging of buffer `cur` complete

    // S(16x64) = Q(16x64) . K^T  : 4 N-tiles x 2 K-steps
    v8f s[4] = {};
#pragma unroll
    for (int jt = 0; jt < 4; ++jt) {
      v16h b0 = frag_row(&Ks[cur][(jt * 16 + (lane & 15)) * 72 + 0], lane);
      v16h b1 = frag_row(&Ks[cur][(jt * 16 + (lane & 15)) * 72 + 32], lane);
      s[jt] = __builtin_amdgcn_wmma_f32_16x16x32_f16(
          false, qF[0], false, b0, (short)0, s[jt], false, false);
      s[jt] = __builtin_amdgcn_wmma_f32_16x16x32_f16(
          false, qF[1], false, b1, (short)0, s[jt], false, false);
    }

    // online softmax (base-2); row = g (+8 for upper half lanes)
    float mp[8];
#pragma unroll
    for (int g = 0; g < 8; ++g) {
      float v0 = fmaxf(fmaxf(s[0][g], s[1][g]), fmaxf(s[2][g], s[3][g]));
      mp[g] = v0 * L2E;
    }
#pragma unroll
    for (int off = 1; off < 16; off <<= 1)
#pragma unroll
      for (int g = 0; g < 8; ++g)
        mp[g] = fmaxf(mp[g], __shfl_xor(mp[g], off, 32));
#pragma unroll
    for (int g = 0; g < 8; ++g) {
      float mnew = fmaxf(mr[g], mp[g]);
      float rsc = exp2f(mr[g] - mnew);
      mr[g] = mnew;
      lr[g] *= rsc;
      o[0][g] *= rsc; o[1][g] *= rsc; o[2][g] *= rsc; o[3][g] *= rsc;
    }

    // P = exp2(S*log2e - m) -> per-wave LDS (C-layout -> A-layout transpose)
    _Float16* pw = &Ps[wave][0];
    const int jc = lane & 15;
#pragma unroll
    for (int jt = 0; jt < 4; ++jt)
#pragma unroll
      for (int g = 0; g < 8; ++g) {
        int rloc = (lane < 16) ? g : g + 8;
        float p = exp2f(s[jt][g] * L2E - mr[g]);
        lr[g] += p;
        pw[rloc * 72 + jt * 16 + jc] = (_Float16)p;
      }
    // same-wave DS ops are in-order: safe to read P back without a barrier

    // O(16x64) += P(16x64) . V(64x64): 2 K-steps x 4 N-tiles
#pragma unroll
    for (int ks = 0; ks < 2; ++ks) {
      v16h pF = frag_row(&pw[(lane & 15) * 72 + ks * 32], lane);
#pragma unroll
      for (int dt = 0; dt < 4; ++dt) {
        v16h vF = frag_row(&Vs[cur][(dt * 16 + (lane & 15)) * 72 + ks * 32], lane);
        o[dt] = __builtin_amdgcn_wmma_f32_16x16x32_f16(
            false, pF, false, vF, (short)0, o[dt], false, false);
      }
    }
    __syncthreads();   // all waves done reading buffer `cur`; it may be
                       // re-targeted by async writes next iteration
  }

  // row sums across the 16 N-lanes, normalize, store f16 [b,h,n,d]
#pragma unroll
  for (int off = 1; off < 16; off <<= 1)
#pragma unroll
    for (int g = 0; g < 8; ++g)
      lr[g] += __shfl_xor(lr[g], off, 32);
  _Float16* ob = ows + (size_t)bh * SEQ * DHEAD;
#pragma unroll
  for (int dt = 0; dt < 4; ++dt) {
    int d = dt * 16 + (lane & 15);
#pragma unroll
    for (int g = 0; g < 8; ++g) {
      int rloc = (lane < 16) ? g : g + 8;
      int n = n0 + wave * 16 + rloc;
      ob[(size_t)n * DHEAD + d] = (_Float16)(o[dt][g] / lr[g]);
    }
  }
}

// ---------------------------------------------------------------- GEMM2: out
__global__ __launch_bounds__(256) void k_out(const _Float16* __restrict__ aws,
                                             const float* __restrict__ wout,
                                             const float* __restrict__ bout,
                                             float* __restrict__ out) {
  __shared__ _Float16 As[64 * 40];
  __shared__ _Float16 Bs[128 * 40];
  const int t = threadIdx.x, lane = t & 31, wave = t >> 5;
  const int wm = wave >> 2, wn = wave & 3;
  const int m0 = blockIdx.y * 64, n0 = blockIdx.x * 128;
  v8f acc[2][2] = {};

  for (int k0 = 0; k0 < DIM; k0 += 32) {
    const int h = k0 >> 6, d0 = k0 & 63;   // BK=32 never crosses a head
    if (k0 + 32 < DIM)
      __builtin_prefetch(wout + (size_t)(k0 + 32 + (t >> 5)) * DIM + n0 + (t & 31) * 4, 0, 1);
    {  // A: 64x32 f16 from [b,h,n,d] workspace (1 v8h per thread)
      int row = t >> 2, c8 = (t & 3) * 8;
      int r = m0 + row, b = r >> 11, n = r & 2047;
      *(v8h*)&As[row * 40 + c8] = *(const v8h*)(
          aws + (((size_t)(b * HEADS + h)) * SEQ + n) * DHEAD + d0 + c8);
    }
#pragma unroll
    for (int i = 0; i < 4; ++i) {  // B: 32x128 fp32 -> transposed f16 LDS
      int v = t + 256 * i;
      int kr = v >> 5, c4 = (v & 31) * 4;
      v4f f = *(const v4f*)(wout + (size_t)(k0 + kr) * DIM + n0 + c4);
      Bs[(c4 + 0) * 40 + kr] = (_Float16)f.x;
      Bs[(c4 + 1) * 40 + kr] = (_Float16)f.y;
      Bs[(c4 + 2) * 40 + kr] = (_Float16)f.z;
      Bs[(c4 + 3) * 40 + kr] = (_Float16)f.w;
    }
    __syncthreads();
    v16h aF[2], bF[2];
#pragma unroll
    for (int mi = 0; mi < 2; ++mi)
      aF[mi] = frag_row(&As[(wm * 32 + mi * 16 + (lane & 15)) * 40], lane);
#pragma unroll
    for (int ni = 0; ni < 2; ++ni)
      bF[ni] = frag_row(&Bs[(wn * 32 + ni * 16 + (lane & 15)) * 40], lane);
#pragma unroll
    for (int mi = 0; mi < 2; ++mi)
#pragma unroll
      for (int ni = 0; ni < 2; ++ni)
        acc[mi][ni] = __builtin_amdgcn_wmma_f32_16x16x32_f16(
            false, aF[mi], false, bF[ni], (short)0, acc[mi][ni], false, false);
    __syncthreads();
  }

#pragma unroll
  for (int mi = 0; mi < 2; ++mi)
#pragma unroll
    for (int ni = 0; ni < 2; ++ni) {
      int c = n0 + wn * 32 + ni * 16 + (lane & 15);
      float bo = bout[c];
#pragma unroll
      for (int g = 0; g < 8; ++g) {
        int r = m0 + wm * 32 + mi * 16 + ((lane < 16) ? g : g + 8);
        out[(size_t)r * DIM + c] = acc[mi][ni][g] + bo;
      }
    }
}

extern "C" void kernel_launch(void* const* d_in, const int* in_sizes, int n_in,
                              void* d_out, int out_size, void* d_ws,
                              size_t ws_size, hipStream_t stream) {
  const float* x    = (const float*)d_in[0];
  const float* wqkv = (const float*)d_in[1];
  const float* wout = (const float*)d_in[2];
  const float* bout = (const float*)d_in[3];
  float* out = (float*)d_out;

  const size_t plane = (size_t)M_TOT * DIM;     // 8192*1024 halves = 16 MB
  _Float16* q  = (_Float16*)d_ws;
  _Float16* k  = q + plane;
  _Float16* v  = k + plane;
  _Float16* ao = v + plane;                     // total ws use: 64 MB

  dim3 g1(NQKV / 128, M_TOT / 64);              // 24 x 128
  k_qkv<<<g1, 256, 0, stream>>>(x, wqkv, q, k, v);

  dim3 g2(SEQ / 64, BATCH * HEADS);             // 32 x 64
  k_attn<<<g2, 128, 0, stream>>>(q, k, v, ao);

  dim3 g3(DIM / 128, M_TOT / 64);               // 8 x 128
  k_out<<<g3, 256, 0, stream>>>(ao, wout, bout, out);
}
